// JaCDEManualJax_8254927143445
// MI455X (gfx1250) — matrix operand order repaired and checked by reading, not compiled, not run on hardware
//
#include <hip/hip_runtime.h>
#include <hip/hip_bf16.h>

// Fused NODE-JVP chain for MI455X (gfx1250, wave32, WMMA).
// B=2048, H=256, IN=64, K_TERMS=8. All GEMMs run as v_wmma_f32_16x16x32_bf16
// with fp32 accumulation; weights live bf16 in LDS (305KB of 320KB/WGP).
// Dual-accumulator tile pairs + double-buffered B fragments hide XDL/LDS
// latency at 1 wave/SIMD occupancy; hardware v_tanh_f32 for the activation.

namespace {

constexpr int B_DIM  = 2048;
constexpr int H_DIM  = 256;
constexpr int IN_DIM = 64;
constexpr int KTERMS = 8;

constexpr int NWAVES         = 4;
constexpr int BLOCK          = 32 * NWAVES;              // 128 threads
constexpr int ROWS_PER_WAVE  = 16;
constexpr int ROWS_PER_BLOCK = NWAVES * ROWS_PER_WAVE;   // 64
constexpr int NBLOCKS        = B_DIM / ROWS_PER_BLOCK;   // 32

constexpr int WPITCH = 264;  // 256 + 8 bf16 pad -> 528B rows, 16B aligned
constexpr int XPITCH = 72;   // 64 + 8
constexpr int SPITCH = 40;   // 32 + 8 (per-wave staging tile pitch)

typedef __bf16 bf16_t;
typedef __attribute__((ext_vector_type(4)))  bf16_t v4bf;
typedef __attribute__((ext_vector_type(8)))  bf16_t v8bf;
typedef __attribute__((ext_vector_type(16))) bf16_t v16bf;
typedef __attribute__((ext_vector_type(4)))  float  v4f;
typedef __attribute__((ext_vector_type(8)))  float  v8f;

__device__ __forceinline__ v8f wmma_bf16(v16bf a, v16bf b, v8f c) {
  // (neg_a, A, neg_b, B, c_mod, C, reuse_a, reuse_b)
  return __builtin_amdgcn_wmma_f32_16x16x32_bf16(false, a, false, b, (short)0, c,
                                                 false, false);
}

// CDNA5 hardware transcendental; v_nops cover the TRANS-op result hazard the
// compiler cannot see through inline asm.
__device__ __forceinline__ float fast_tanh(float x) {
  float r;
  asm volatile("v_tanh_f32 %0, %1\n\tv_nop\n\tv_nop" : "=v"(r) : "v"(x));
  return r;
}

// B-fragment (32x16 bf16) from row-major W[n][k] in LDS.
// Lane l: n = l&15, g = l>>4; VGPR v: k = (v>=4)*16 + g*8 + (v&3)*2.
__device__ __forceinline__ v16bf load_b_frag(const bf16_t* W, int pitch, int nt,
                                             int kt, int lane) {
  const int n = lane & 15, g = lane >> 4;
  const bf16_t* p = W + (nt * 16 + n) * pitch + kt * 32 + g * 8;
  v8bf lo = *(const v8bf*)(p);      // VGPRs 0..3
  v8bf hi = *(const v8bf*)(p + 16); // VGPRs 4..7
  return __builtin_shufflevector(lo, hi, 0, 1, 2, 3, 4, 5, 6, 7,
                                 8, 9, 10, 11, 12, 13, 14, 15);
}

// A-fragment (16x32 bf16) from a [16][SPITCH] row-major staging tile in LDS.
// Lane l: m = l&15, g = l>>4; VGPR v: k = (v>=4)*16 + g*8 + (v&3)*2.
__device__ __forceinline__ v16bf load_a_stage(const bf16_t* S, int lane) {
  const int m = lane & 15, g = lane >> 4;
  const bf16_t* p = S + m * SPITCH + g * 8;
  v8bf lo = *(const v8bf*)(p);       // k = g*8 .. g*8+7  -> VGPRs 0..3
  v8bf hi = *(const v8bf*)(p + 16);  // k = 16+g*8 ..     -> VGPRs 4..7
  return __builtin_shufflevector(lo, hi, 0, 1, 2, 3, 4, 5, 6, 7,
                                 8, 9, 10, 11, 12, 13, 14, 15);
}

// A-fragment straight from global fp32 row-major [16 rows][stride].
__device__ __forceinline__ v16bf load_a_global(const float* __restrict__ rows,
                                               int stride, int kbase, int lane) {
  const int m = lane & 15, g = lane >> 4;
  const float* row = rows + m * stride;
  v16bf a;
#pragma unroll
  for (int v = 0; v < 8; ++v) {
    const int k = kbase + ((v & 4) << 2) + g * 8 + (v & 3) * 2;
    a[2 * v]     = (bf16_t)row[k];
    a[2 * v + 1] = (bf16_t)row[k + 1];
  }
  return a;
}

// Two output tiles (nt0, nt0+1) share the A operand: two independent WMMA
// accumulation chains (hides XDL latency) with double-buffered B fragments
// (hides LDS latency; wait becomes dscnt<=2 instead of 0).
template <int KT>
__device__ __forceinline__ void gemm_pair(const v16bf (&A)[KT], const bf16_t* W,
                                          int pitch, int nt0, int lane,
                                          v8f& acc0, v8f& acc1) {
  v16bf bA = load_b_frag(W, pitch, nt0, 0, lane);
  v16bf bB = load_b_frag(W, pitch, nt0 + 1, 0, lane);
#pragma unroll
  for (int kt = 0; kt < KT; ++kt) {
    v16bf nA = bA, nB = bB;
    if (kt + 1 < KT) {
      nA = load_b_frag(W, pitch, nt0, kt + 1, lane);
      nB = load_b_frag(W, pitch, nt0 + 1, kt + 1, lane);
    }
    acc0 = wmma_bf16(A[kt], bA, acc0);
    acc1 = wmma_bf16(A[kt], bB, acc1);
    bA = nA;
    bB = nB;
  }
}

// C-layout slab (two 16x16 f32 tiles) -> bf16 A-fragment via per-wave LDS
// staging. Same-wave DS ops execute in order (ISA 7.3), so no explicit
// s_wait_dscnt is needed between the stores and the read-back; the compiler
// inserts the wait before the loaded registers are consumed.
__device__ __forceinline__ v16bf c2a(v8f c0, v8f c1, bf16_t* stage, int lane) {
  const int n = lane & 15, g = lane >> 4;
#pragma unroll
  for (int r = 0; r < 8; ++r) {
    const int row = r + 8 * g;               // C layout: lane holds [r+8g][n]
    stage[row * SPITCH + n]      = (bf16_t)c0[r];
    stage[row * SPITCH + 16 + n] = (bf16_t)c1[r];
  }
  asm volatile("" ::: "memory");             // keep program order only
  return load_a_stage(stage, lane);
}

__device__ __forceinline__ v8f gate_relu(v8f acc, const unsigned* mask, int nt) {
  const unsigned w = mask[nt >> 2];
#pragma unroll
  for (int r = 0; r < 8; ++r)
    acc[r] = ((w >> (((nt & 3) << 3) + r)) & 1u) ? acc[r] : 0.0f;
  return acc;
}

__device__ __forceinline__ v4bf cvt4(v4f a) {
  v4bf r;
  r[0] = (bf16_t)a[0];
  r[1] = (bf16_t)a[1];
  r[2] = (bf16_t)a[2];
  r[3] = (bf16_t)a[3];
  return r;
}

__global__ __launch_bounds__(BLOCK, 1) void node_jvp_kernel(
    const float* __restrict__ h, const float* __restrict__ x,
    const float* __restrict__ xdot, const float* __restrict__ wx,
    const float* __restrict__ wh, const float* __restrict__ wout,
    const float* __restrict__ b0, const float* __restrict__ b1,
    float* __restrict__ out) {
  __shared__ bf16_t sWH[H_DIM * WPITCH];          // 132 KB
  __shared__ bf16_t sWO[H_DIM * WPITCH];          // 132 KB
  __shared__ bf16_t sWX[H_DIM * XPITCH];          //  36 KB
  __shared__ bf16_t sStage[NWAVES][16 * SPITCH];  //   5 KB

  const int tid  = threadIdx.x;
  const int lane = tid & 31;
  const int wave = tid >> 5;

  // ---- cooperative fp32 -> bf16 weight staging, float4-vectorized ----
  for (int i4 = tid; i4 < (H_DIM * H_DIM) / 4; i4 += BLOCK) {
    const int i = i4 * 4, r = i >> 8, c = i & 255;
    *(v4bf*)(sWH + r * WPITCH + c) = cvt4(((const v4f*)wh)[i4]);
    *(v4bf*)(sWO + r * WPITCH + c) = cvt4(((const v4f*)wout)[i4]);
  }
  for (int i4 = tid; i4 < (H_DIM * IN_DIM) / 4; i4 += BLOCK) {
    const int i = i4 * 4, r = i >> 6, c = i & 63;
    *(v4bf*)(sWX + r * XPITCH + c) = cvt4(((const v4f*)wx)[i4]);
  }
  __syncthreads();

  bf16_t* stage = sStage[wave];
  const int r0 = (blockIdx.x * NWAVES + wave) * ROWS_PER_WAVE;
  const int n  = lane & 15;
  const int g  = lane >> 4;

  // ---- A-fragments of the batch-strip inputs ----
  v16bf Ah[8], Ax[2], Axd[2];
#pragma unroll
  for (int kt = 0; kt < 8; ++kt)
    Ah[kt] = load_a_global(h + r0 * H_DIM, H_DIM, kt * 32, lane);
#pragma unroll
  for (int kt = 0; kt < 2; ++kt) {
    Ax[kt]  = load_a_global(x + r0 * IN_DIM, IN_DIM, kt * 32, lane);
    Axd[kt] = load_a_global(xdot + r0 * IN_DIM, IN_DIM, kt * 32, lane);
  }

  unsigned mask[4] = {0u, 0u, 0u, 0u};  // drelu bitmask: bit = [nt][r]
  v16bf Arelu[8];

  // ---- Pass 1: l1 = x@wx.T + h@wh.T + b0 ; relu ; drelu mask ; -> Arelu ----
#pragma unroll
  for (int j = 0; j < 8; ++j) {
    const int nt0 = 2 * j;
    v8f a0, a1;
    {
      const float bv0 = b0[nt0 * 16 + n], bv1 = b0[(nt0 + 1) * 16 + n];
#pragma unroll
      for (int r = 0; r < 8; ++r) {
        a0[r] = bv0;
        a1[r] = bv1;
      }
    }
    gemm_pair<8>(Ah, sWH, WPITCH, nt0, lane, a0, a1);
    gemm_pair<2>(Ax, sWX, XPITCH, nt0, lane, a0, a1);
#pragma unroll
    for (int t = 0; t < 2; ++t) {
      v8f& a = t ? a1 : a0;
      const int nt = nt0 + t;
#pragma unroll
      for (int r = 0; r < 8; ++r) {
        const bool pos = a[r] > 0.0f;
        mask[nt >> 2] |= (pos ? 1u : 0u) << (((nt & 3) << 3) + r);
        a[r] = pos ? a[r] : 0.0f;
      }
    }
    Arelu[j] = c2a(a0, a1, stage, lane);
  }

  // ---- Pass 2: lout = relu@wout.T + b1 ; dtanh = 1 - tanh^2 ----
  v8f dtv[16];
#pragma unroll
  for (int j = 0; j < 8; ++j) {
    const int nt0 = 2 * j;
    v8f a0, a1;
    {
      const float bv0 = b1[nt0 * 16 + n], bv1 = b1[(nt0 + 1) * 16 + n];
#pragma unroll
      for (int r = 0; r < 8; ++r) {
        a0[r] = bv0;
        a1[r] = bv1;
      }
    }
    gemm_pair<8>(Arelu, sWO, WPITCH, nt0, lane, a0, a1);
#pragma unroll
    for (int r = 0; r < 8; ++r) {
      const float t0 = fast_tanh(a0[r]);
      const float t1 = fast_tanh(a1[r]);
      a0[r] = 1.0f - t0 * t0;
      a1[r] = 1.0f - t1 * t1;
    }
    dtv[nt0]     = a0;
    dtv[nt0 + 1] = a1;
  }

  // ---- Pass 3: u = xdot@wx.T ; gated by drelu -> Agu ----
  v16bf Agu[8];
#pragma unroll
  for (int j = 0; j < 8; ++j) {
    const int nt0 = 2 * j;
    v8f a0 = {}, a1 = {};
    gemm_pair<2>(Axd, sWX, XPITCH, nt0, lane, a0, a1);
    a0 = gate_relu(a0, mask, nt0);
    a1 = gate_relu(a1, mask, nt0 + 1);
    Agu[j] = c2a(a0, a1, stage, lane);
  }

  // ---- Pass 4: jx = dtanh ⊙ (Agu @ wout.T) ; h_dot = v = jx ----
  v8f hdot[16];
  v16bf Av[8];
#pragma unroll
  for (int j = 0; j < 8; ++j) {
    const int nt0 = 2 * j;
    v8f a0 = {}, a1 = {};
    gemm_pair<8>(Agu, sWO, WPITCH, nt0, lane, a0, a1);
#pragma unroll
    for (int r = 0; r < 8; ++r) {
      a0[r] *= dtv[nt0][r];
      a1[r] *= dtv[nt0 + 1][r];
    }
    hdot[nt0]     = a0;
    hdot[nt0 + 1] = a1;
    Av[j] = c2a(a0, a1, stage, lane);
  }

  // ---- Neumann-series loop: v = dtanh ⊙ ((drelu ⊙ (v@wh.T)) @ wout.T) ----
  for (int it = 0; it < KTERMS; ++it) {
    v16bf At[8];
#pragma unroll
    for (int j = 0; j < 8; ++j) {
      const int nt0 = 2 * j;
      v8f a0 = {}, a1 = {};
      gemm_pair<8>(Av, sWH, WPITCH, nt0, lane, a0, a1);
      a0 = gate_relu(a0, mask, nt0);
      a1 = gate_relu(a1, mask, nt0 + 1);
      At[j] = c2a(a0, a1, stage, lane);
    }
#pragma unroll
    for (int j = 0; j < 8; ++j) {
      const int nt0 = 2 * j;
      v8f a0 = {}, a1 = {};
      gemm_pair<8>(At, sWO, WPITCH, nt0, lane, a0, a1);
#pragma unroll
      for (int r = 0; r < 8; ++r) {
        a0[r] *= dtv[nt0][r];
        a1[r] *= dtv[nt0 + 1][r];
        hdot[nt0][r]     += a0[r];
        hdot[nt0 + 1][r] += a1[r];
      }
      Av[j] = c2a(a0, a1, stage, lane);
    }
  }

  // ---- write h_dot ----
#pragma unroll
  for (int nt = 0; nt < 16; ++nt)
#pragma unroll
    for (int r = 0; r < 8; ++r)
      out[(r0 + r + 8 * g) * H_DIM + nt * 16 + n] = hdot[nt][r];
}

}  // namespace

extern "C" void kernel_launch(void* const* d_in, const int* in_sizes, int n_in,
                              void* d_out, int out_size, void* d_ws,
                              size_t ws_size, hipStream_t stream) {
  (void)in_sizes; (void)n_in; (void)out_size; (void)d_ws; (void)ws_size;
  const float* h    = (const float*)d_in[0];
  const float* x    = (const float*)d_in[1];
  const float* xdot = (const float*)d_in[2];
  const float* wx   = (const float*)d_in[3];
  const float* wh   = (const float*)d_in[4];
  const float* wout = (const float*)d_in[5];
  const float* b0   = (const float*)d_in[6];
  const float* b1   = (const float*)d_in[7];
  float* out        = (float*)d_out;

  node_jvp_kernel<<<dim3(NBLOCKS), dim3(BLOCK), 0, stream>>>(
      h, x, xdot, wx, wh, wout, b0, b1, out);
}
